// MultiTaskCGCNN_90486370993094
// MI455X (gfx1250) — compile-verified, matrix operand order checked by reading, not run on hardware
//
#include <hip/hip_runtime.h>
#include <hip/hip_bf16.h>

// ---------------------------------------------------------------------------
// MultiTask CGCNN for MI455X (gfx1250).
// Core: edge-gate MLPs of conv2/conv3 cast as bf16 WMMA GEMMs
//   [1.6M x 288] x [288 x 128]  (x2 matrices x2 convs) with fp32 accumulate,
// fused sigmoid*softplus gate and atomic scatter-add into the residual buffer.
// v2: each wave processes 2 edge-subtiles per B-fragment load (R_B=2),
//     halving packed-weight refetch traffic from L0/L2.
// ---------------------------------------------------------------------------

typedef __attribute__((ext_vector_type(16))) __bf16 v16bf;
typedef __attribute__((ext_vector_type(8)))  __bf16 v8bf;
typedef __attribute__((ext_vector_type(8)))  float  v8f;
typedef unsigned short u16;
typedef unsigned int   u32;

#define N_NODES  50000
#define N_EDGES  1600000
#define HID      128
#define KDIM     288          // 2*HID + 32
#define N_GRAPHS 512
#define TILE_E   64
#define ZSTR     296          // padded LDS row stride in halfs (288 + 8)

__device__ __forceinline__ u16 f2bf(float f) {
    u32 u = __float_as_uint(f);
    u32 r = u + 0x7FFFu + ((u >> 16) & 1u);   // round-to-nearest-even
    return (u16)(r >> 16);
}

// ---------------------------------------------------------------- utilities
__global__ void k_copy_f(const float* __restrict__ a, float* __restrict__ b, int n) {
    int i = blockIdx.x * 256 + threadIdx.x;
    if (i < n) b[i] = a[i];
}

__global__ void k_cvt_bf(const float* __restrict__ a, u16* __restrict__ b, int n) {
    int i = blockIdx.x * 256 + threadIdx.x;
    if (i < n) b[i] = f2bf(a[i]);
}

__global__ void k_relu_cvt(const float* __restrict__ d, float* __restrict__ hf,
                           u16* __restrict__ hb, int n) {
    int i = blockIdx.x * 256 + threadIdx.x;
    if (i < n) {
        float v = fmaxf(d[i], 0.0f);
        hf[i] = v;
        hb[i] = f2bf(v);
    }
}

// Pack W[288,128] (row-major k,n; fp32) into bf16 B-fragment image:
// P[kt][nt][lane][16], lane<16 -> n=nt*16+lane, lane>=16 same n, kbase=8.
// half i: k = kt*32 + kbase + (i<8 ? i : 16+(i-8)).
__global__ void k_pack_w(const float* __restrict__ W, u16* __restrict__ P) {
    int idx = blockIdx.x * 256 + threadIdx.x;     // 9*8*32*16 = 36864
    if (idx >= 9 * 8 * 32 * 16) return;
    int i    = idx & 15;
    int lane = (idx >> 4) & 31;
    int nt   = (idx >> 9) & 7;
    int kt   = idx >> 12;
    int n = nt * 16 + (lane & 15);
    int k = kt * 32 + ((lane >> 4) * 8) + ((i < 8) ? i : (16 + (i - 8)));
    P[idx] = f2bf(W[k * HID + n]);
}

// ---------------------------------------------------------------- conv1 (C=3)
__global__ __launch_bounds__(256)
void k_conv1_edge(const float* __restrict__ x, const float* __restrict__ ea,
                  const int* __restrict__ src, const int* __restrict__ dst,
                  const float* __restrict__ W1f, const float* __restrict__ b1f,
                  const float* __restrict__ W1s, const float* __restrict__ b1s,
                  float* __restrict__ h1) {
    __shared__ float wf[38 * 3], ws[38 * 3], bfb[3], bsb[3];
    int t = threadIdx.x;
    if (t < 114) { wf[t] = W1f[t]; ws[t] = W1s[t]; }
    if (t < 3)   { bfb[t] = b1f[t]; bsb[t] = b1s[t]; }
    __syncthreads();
    int e = blockIdx.x * 256 + t;
    if (e >= N_EDGES) return;
    int d = dst[e], s = src[e];
    float z[38];
    z[0] = x[d * 3 + 0]; z[1] = x[d * 3 + 1]; z[2] = x[d * 3 + 2];
    z[3] = x[s * 3 + 0]; z[4] = x[s * 3 + 1]; z[5] = x[s * 3 + 2];
#pragma unroll
    for (int i = 0; i < 32; ++i) z[6 + i] = ea[e * 32 + i];
#pragma unroll
    for (int c = 0; c < 3; ++c) {
        float af = bfb[c], as = bsb[c];
#pragma unroll
        for (int k = 0; k < 38; ++k) { af += z[k] * wf[k * 3 + c]; as += z[k] * ws[k * 3 + c]; }
        float sig = 1.0f / (1.0f + __expf(-af));
        float sp  = (as > 15.0f) ? as : log1pf(__expf(as));
        atomicAdd(&h1[d * 3 + c], sig * sp);
    }
}

// node_proj: h1[N,3] @ Wp[3,128] + bp, relu -> hf32 + hbf16
__global__ void k_nodeproj(const float* __restrict__ h1, const float* __restrict__ Wp,
                           const float* __restrict__ bp, float* __restrict__ hf,
                           u16* __restrict__ hb) {
    int idx = blockIdx.x * 256 + threadIdx.x;
    if (idx >= N_NODES * HID) return;
    int n = idx >> 7, c = idx & 127;
    float v = bp[c] + h1[n * 3 + 0] * Wp[c]
                    + h1[n * 3 + 1] * Wp[HID + c]
                    + h1[n * 3 + 2] * Wp[2 * HID + c];
    v = fmaxf(v, 0.0f);
    hf[idx] = v;
    hb[idx] = f2bf(v);
}

// --------------------------------------------------- edge-gate WMMA GEMM core
// One block = 64 edges, 128 threads (4 waves).
// Wave w: edge subtile-pair sp=w>>1 (32 edges), column half ch2=w&1 (64 cols).
// Each B-fragment load feeds 2 WMMAs (both subtiles); A frags reused 8x.
__global__ __launch_bounds__(128)
void k_edge_gate_gemm(const u16* __restrict__ hbf, const u16* __restrict__ ebf,
                      const int* __restrict__ src, const int* __restrict__ dst,
                      const u16* __restrict__ packF, const u16* __restrict__ packS,
                      const float* __restrict__ bF, const float* __restrict__ bS,
                      float* __restrict__ out) {
    __shared__ u16 zt[TILE_E * ZSTR];     // 37,888 B
    __shared__ int dsts[TILE_E];

    const int e0 = blockIdx.x * TILE_E;
    const int t  = threadIdx.x;

    if (t < TILE_E) dsts[t] = dst[e0 + t];

    // Stage Z = [h[dst](128) | h[src](128) | e(32)] as bf16.
    // 2 threads per edge, each moves 18 chunks of 8 halfs (16 B) b128 ld/st.
    {
        const int e  = t >> 1;
        const int s2 = t & 1;
        const int ge = e0 + e;
        const int d  = dst[ge];
        const int sn = src[ge];
        u16* zr = &zt[e * ZSTR];
#pragma unroll
        for (int j = 0; j < 18; ++j) {
            int ch = s2 * 18 + j;                   // 0..35
            const u16* gp;
            if (ch < 16)      gp = &hbf[(size_t)d  * HID + ch * 8];
            else if (ch < 32) gp = &hbf[(size_t)sn * HID + (ch - 16) * 8];
            else              gp = &ebf[(size_t)ge * 32  + (ch - 32) * 8];
            *(uint4*)(&zr[ch * 8]) = *(const uint4*)gp;
        }
    }
    __syncthreads();

    const int w    = t >> 5;                 // wave 0..3
    const int lane = t & 31;
    const int sp   = w >> 1;                 // subtile pair 0..1 (edges sp*32..sp*32+31)
    const int ch2  = w & 1;                  // column half
    const int row  = lane & 15;              // M (edge) for A, N (col) for D
    const int kb   = (lane >> 4) * 8;

    // Preload A fragments for both subtiles: 2 x 9 ktiles.
    v16bf a0[9], a1[9];
    {
        const u16* zr0 = &zt[(sp * 32 + row) * ZSTR];
        const u16* zr1 = &zt[(sp * 32 + 16 + row) * ZSTR];
#pragma unroll
        for (int kt = 0; kt < 9; ++kt) {
            int base = kt * 32 + kb;
            v8bf lo0 = *(const v8bf*)(&zr0[base]);
            v8bf hi0 = *(const v8bf*)(&zr0[base + 16]);
            v8bf lo1 = *(const v8bf*)(&zr1[base]);
            v8bf hi1 = *(const v8bf*)(&zr1[base + 16]);
            v16bf av0, av1;
#pragma unroll
            for (int i = 0; i < 8; ++i) {
                av0[i] = lo0[i]; av0[i + 8] = hi0[i];
                av1[i] = lo1[i]; av1[i + 8] = hi1[i];
            }
            a0[kt] = av0;
            a1[kt] = av1;
        }
    }

    const int mbase = (lane >> 4) * 8;
#pragma unroll
    for (int nt = 0; nt < 4; ++nt) {
        const int ntg = ch2 * 4 + nt;
        const u16* pF = &packF[(ntg * 32 + lane) * 16];
        const u16* pS = &packS[(ntg * 32 + lane) * 16];
        v8f accF0 = {}, accS0 = {}, accF1 = {}, accS1 = {};
#pragma unroll
        for (int kt = 0; kt < 9; ++kt) {
            v16bf bf = *(const v16bf*)(pF + kt * (8 * 32 * 16));
            v16bf bs = *(const v16bf*)(pS + kt * (8 * 32 * 16));
            accF0 = __builtin_amdgcn_wmma_f32_16x16x32_bf16(false, a0[kt], false, bf,
                                                            (short)0, accF0, false, false);
            accF1 = __builtin_amdgcn_wmma_f32_16x16x32_bf16(false, a1[kt], false, bf,
                                                            (short)0, accF1, false, false);
            accS0 = __builtin_amdgcn_wmma_f32_16x16x32_bf16(false, a0[kt], false, bs,
                                                            (short)0, accS0, false, false);
            accS1 = __builtin_amdgcn_wmma_f32_16x16x32_bf16(false, a1[kt], false, bs,
                                                            (short)0, accS1, false, false);
        }
        const int col = ch2 * 64 + nt * 16 + row;
        const float bfb = bF[col], bsb = bS[col];
#pragma unroll
        for (int j = 0; j < 8; ++j) {
            int m = mbase + j;
            {   // subtile 0 of the pair
                float xf = accF0[j] + bfb;
                float xs = accS0[j] + bsb;
                float sig = 1.0f / (1.0f + __expf(-xf));
                float sp_ = (xs > 15.0f) ? xs : log1pf(__expf(xs));
                int dn = dsts[sp * 32 + m];
                atomicAdd(&out[(size_t)dn * HID + col], sig * sp_);
            }
            {   // subtile 1 of the pair
                float xf = accF1[j] + bfb;
                float xs = accS1[j] + bsb;
                float sig = 1.0f / (1.0f + __expf(-xf));
                float sp_ = (xs > 15.0f) ? xs : log1pf(__expf(xs));
                int dn = dsts[sp * 32 + 16 + m];
                atomicAdd(&out[(size_t)dn * HID + col], sig * sp_);
            }
        }
    }
}

// ---------------------------------------------------------------- pool + head
__global__ void k_pool_zero(float* __restrict__ g, float* __restrict__ cnt) {
    int i = blockIdx.x * 256 + threadIdx.x;
    if (i < N_GRAPHS * HID) g[i] = 0.0f;
    if (i < N_GRAPHS) cnt[i] = 0.0f;
}

__global__ void k_pool_acc(const float* __restrict__ D, const int* __restrict__ batch,
                           float* __restrict__ g, float* __restrict__ cnt) {
    int idx = blockIdx.x * 256 + threadIdx.x;
    if (idx >= N_NODES * HID) return;
    int n = idx >> 7, c = idx & 127;
    int b = batch[n];
    atomicAdd(&g[b * HID + c], fmaxf(D[idx], 0.0f));   // relu fused into pool
    if (c == 0) atomicAdd(&cnt[b], 1.0f);
}

__global__ __launch_bounds__(128)
void k_head(const float* __restrict__ g, const float* __restrict__ cnt,
            const float* __restrict__ Wfc, const float* __restrict__ bfc,
            const float* __restrict__ Wh, const float* __restrict__ bh,
            float* __restrict__ out) {
    __shared__ float gr[HID], fc[HID];
    int b = blockIdx.x, c = threadIdx.x;
    float cn = fmaxf(cnt[b], 1.0f);
    gr[c] = g[b * HID + c] / cn;
    __syncthreads();
    float acc = bfc[c];
    for (int k = 0; k < HID; ++k) acc += gr[k] * Wfc[k * HID + c];
    fc[c] = fmaxf(acc, 0.0f);
    __syncthreads();
    if (c < 5) {
        float a = bh[c];
        for (int k = 0; k < HID; ++k) a += fc[k] * Wh[k * 5 + c];
        out[b * 5 + c] = a;
    }
}

// ---------------------------------------------------------------------------
extern "C" void kernel_launch(void* const* d_in, const int* in_sizes, int n_in,
                              void* d_out, int out_size, void* d_ws, size_t ws_size,
                              hipStream_t stream) {
    const float* x    = (const float*)d_in[0];
    const float* ea   = (const float*)d_in[1];
    const int*   eidx = (const int*)d_in[2];
    const int*   src  = eidx;
    const int*   dst  = eidx + N_EDGES;
    const int*   batch = (const int*)d_in[3];
    const float* W1f = (const float*)d_in[4];
    const float* b1f = (const float*)d_in[5];
    const float* W1s = (const float*)d_in[6];
    const float* b1s = (const float*)d_in[7];
    const float* Wp  = (const float*)d_in[8];
    const float* bp  = (const float*)d_in[9];
    const float* W2f = (const float*)d_in[10];
    const float* b2f = (const float*)d_in[11];
    const float* W2s = (const float*)d_in[12];
    const float* b2s = (const float*)d_in[13];
    const float* W3f = (const float*)d_in[14];
    const float* b3f = (const float*)d_in[15];
    const float* W3s = (const float*)d_in[16];
    const float* b3s = (const float*)d_in[17];
    const float* Wfc = (const float*)d_in[18];
    const float* bfc = (const float*)d_in[19];
    const float* Wh  = (const float*)d_in[20];
    const float* bh  = (const float*)d_in[21];
    float* out = (float*)d_out;

    // -------- workspace carve (256 B aligned) --------
    char* p = (char*)d_ws;
    auto take = [&](size_t bytes) {
        char* r = p;
        p += (bytes + 255) & ~(size_t)255;
        return r;
    };
    u16*   ebf   = (u16*)  take((size_t)N_EDGES * 32 * 2);     // 102.4 MB
    u16*   hbf   = (u16*)  take((size_t)N_NODES * HID * 2);    // 12.8 MB
    float* hf    = (float*)take((size_t)N_NODES * HID * 4);    // 25.6 MB
    float* Dacc  = (float*)take((size_t)N_NODES * HID * 4);    // 25.6 MB
    float* h1    = (float*)take((size_t)N_NODES * 3 * 4);      // 0.6 MB
    u16*   p2f   = (u16*)  take((size_t)KDIM * HID * 2);
    u16*   p2s   = (u16*)  take((size_t)KDIM * HID * 2);
    u16*   p3f   = (u16*)  take((size_t)KDIM * HID * 2);
    u16*   p3s   = (u16*)  take((size_t)KDIM * HID * 2);
    float* gpool = (float*)take((size_t)N_GRAPHS * HID * 4);
    float* cnt   = (float*)take((size_t)N_GRAPHS * 4);
    (void)ws_size; (void)n_in; (void)in_sizes; (void)out_size;

    const int NH = N_NODES * HID;                       // 6,400,000
    const int gNH = NH / 256;                           // 25,000
    const int gTiles = N_EDGES / TILE_E;                // 25,000

    // ---- conv1 (C=3): h1 = x + scatter(gate) ----
    k_copy_f<<<(N_NODES * 3 + 255) / 256, 256, 0, stream>>>(x, h1, N_NODES * 3);
    k_conv1_edge<<<N_EDGES / 256, 256, 0, stream>>>(x, ea, src, dst, W1f, b1f, W1s, b1s, h1);

    // ---- node_proj -> hf32 / hbf16 ----
    k_nodeproj<<<gNH, 256, 0, stream>>>(h1, Wp, bp, hf, hbf);

    // ---- one-time bf16 converts / weight packs ----
    k_cvt_bf<<<(N_EDGES * 32) / 256, 256, 0, stream>>>(ea, ebf, N_EDGES * 32);
    k_pack_w<<<144, 256, 0, stream>>>(W2f, p2f);
    k_pack_w<<<144, 256, 0, stream>>>(W2s, p2s);
    k_pack_w<<<144, 256, 0, stream>>>(W3f, p3f);
    k_pack_w<<<144, 256, 0, stream>>>(W3s, p3s);

    // ---- conv2: Dacc = hf (residual) + scatter(gate); relu -> hf/hbf ----
    k_copy_f<<<gNH, 256, 0, stream>>>(hf, Dacc, NH);
    k_edge_gate_gemm<<<gTiles, 128, 0, stream>>>(hbf, ebf, src, dst, p2f, p2s, b2f, b2s, Dacc);
    k_relu_cvt<<<gNH, 256, 0, stream>>>(Dacc, hf, hbf, NH);

    // ---- conv3 ----
    k_copy_f<<<gNH, 256, 0, stream>>>(hf, Dacc, NH);
    k_edge_gate_gemm<<<gTiles, 128, 0, stream>>>(hbf, ebf, src, dst, p3f, p3s, b3f, b3s, Dacc);

    // ---- global mean pool (relu fused) + fc + heads ----
    k_pool_zero<<<(N_GRAPHS * HID) / 256, 256, 0, stream>>>(gpool, cnt);
    k_pool_acc<<<gNH, 256, 0, stream>>>(Dacc, batch, gpool, cnt);
    k_head<<<N_GRAPHS, 128, 0, stream>>>(gpool, cnt, Wfc, bfc, Wh, bh, out);
}
